// Level2Quantizer_80616536146014
// MI455X (gfx1250) — compile-verified, matrix operand order checked by reading, not run on hardware
//
#include <hip/hip_runtime.h>
#include <hip/hip_bf16.h>

#define BB   32
#define TT   2048
#define DD   1024
#define BDIM 256
#define KK2  64

typedef __attribute__((ext_vector_type(16))) __bf16 v16bf;
typedef __attribute__((ext_vector_type(8)))  float  v8f;

union Frag {
  unsigned u[8];
  uint4    q[2];
  v16bf    v;
};

__device__ __forceinline__ unsigned short bf16r(float a) {
  unsigned ua = __builtin_bit_cast(unsigned, a);
  ua = (ua + 0x7FFFu + ((ua >> 16) & 1u)) >> 16;   // round-to-nearest-even
  return (unsigned short)ua;
}
__device__ __forceinline__ unsigned pack2(float a, float b) {
  return (unsigned)bf16r(a) | ((unsigned)bf16r(b) << 16);
}
__device__ __forceinline__ v8f zero8() {
  v8f z;
#pragma unroll
  for (int i = 0; i < 8; ++i) z[i] = 0.f;
  return z;
}

// ---- W1 (D,BD) -> bf16, K-step-tiled: w1t[step][n][k32], step=k/32 ----
__global__ void convert_w1(const float* __restrict__ W1, unsigned short* __restrict__ w1t) {
  int idx = blockIdx.x * blockDim.x + threadIdx.x;       // BD*D total
  int step = idx / (BDIM * 32);
  int rem  = idx % (BDIM * 32);
  int n = rem / 32, k = rem % 32;
  w1t[idx] = bf16r(W1[(size_t)(step * 32 + k) * BDIM + n]);
}
// ---- W2 (BD,D) -> bf16, K-step-tiled: w2t[step][n over D][k32] ----
__global__ void convert_w2(const float* __restrict__ W2, unsigned short* __restrict__ w2t) {
  int idx = blockIdx.x * blockDim.x + threadIdx.x;       // D*BD total
  int step = idx / (DD * 32);
  int rem  = idx % (DD * 32);
  int n = rem / 32, k = rem % 32;
  w2t[idx] = bf16r(W2[(size_t)(step * 32 + k) * DD + n]);
}

// ---- gather per-batch codebook; normalized bf16 copy in K-step-tiled layout ----
__global__ void prep_cb(const float* __restrict__ codebooks, const long long* __restrict__ l1,
                        float* __restrict__ cbf, unsigned short* __restrict__ cbn) {
  int b = blockIdx.x;
  int lane = threadIdx.x & 31, wave = threadIdx.x >> 5;
  long long li = l1[b];
  const float* src = codebooks + (size_t)li * KK2 * BDIM;
  for (int cc = 0; cc < 8; ++cc) {
    int c = wave * 8 + cc;
    const float* row = src + (size_t)c * BDIM;
    float vals[8]; float ss = 0.f;
#pragma unroll
    for (int i = 0; i < 8; ++i) { float v = row[lane + i * 32]; vals[i] = v; ss += v * v; }
#pragma unroll
    for (int m = 1; m < 32; m <<= 1) ss += __shfl_xor(ss, m, 32);
    float inv = 1.f / fmaxf(sqrtf(ss), 1e-12f);
#pragma unroll
    for (int i = 0; i < 8; ++i) {
      int e = lane + i * 32;
      cbf[((size_t)b * KK2 + c) * BDIM + e] = vals[i];
      // tiled: [b][step=i][c][lane]
      cbn[(size_t)b * KK2 * BDIM + i * (KK2 * 32) + c * 32 + lane] = bf16r(vals[i] * inv);
    }
  }
}

// ---- fused: GEMM1 + LN + L2 + logits WMMA + softmax/argmax + emb_low gather ----
__global__ __launch_bounds__(256) void bottleneck(
    const float* __restrict__ x, const unsigned* __restrict__ w1t32,
    const float* __restrict__ b1, const float* __restrict__ g1, const float* __restrict__ be1,
    const float* __restrict__ cbf, const unsigned* __restrict__ cbn32,
    float* __restrict__ out_hard, float* __restrict__ out_soft, float* __restrict__ out_low) {
  // phase A: xs[2] (2x512 u32) + w1s[2] (2x4096 u32) = 9216 u32
  // phase B: hs 8192 f32 | hn 4096 u32 | lg 2048 f32 = 14336 u32 (56 KB, aliased)
  __shared__ __align__(16) unsigned smem[14336];

  int tid = threadIdx.x, lane = tid & 31, wave = tid >> 5;
  int rowTile = wave >> 2, cg = wave & 3;
  int m = lane & 15, hh = lane >> 4;
  size_t row0 = (size_t)blockIdx.x * 32;
  int bIdx = (int)(row0 / TT);
  int ldRow = tid >> 3, ldSeg = tid & 7;

  v8f acc0 = zero8(), acc1 = zero8(), acc2 = zero8(), acc3 = zero8();

  const float* xbase  = x + (row0 + ldRow) * DD + ldSeg * 4;
  const uint4* wqbase = (const uint4*)w1t32 + tid * 4;   // tile stride = 1024 uint4

  // named-scalar register pipeline (no arrays / lambdas -> stays in VGPRs)
  float4 xr = *(const float4*)xbase;
  uint4  w0 = wqbase[0], w1 = wqbase[1], w2 = wqbase[2], w3 = wqbase[3];

  for (int step = 0; step < 32; ++step) {
    int buf = step & 1;
    {
      unsigned* xs  = smem + buf * 512;
      unsigned* wls = smem + 1024 + buf * 4096;
      xs[ldRow * 16 + ldSeg * 2]     = pack2(xr.x, xr.y);
      xs[ldRow * 16 + ldSeg * 2 + 1] = pack2(xr.z, xr.w);
      uint4* d = (uint4*)(wls + tid * 16);
      d[0] = w0; d[1] = w1; d[2] = w2; d[3] = w3;
    }
    __syncthreads();
    if (step + 1 < 32) {
      xr = *(const float4*)(xbase + (step + 1) * 32);
      const uint4* g = wqbase + (size_t)(step + 1) * 1024;
      w0 = g[0]; w1 = g[1]; w2 = g[2]; w3 = g[3];
      if (step + 2 < 32)
        __builtin_prefetch(xbase + (step + 2) * 32, 0, 1);
    }
    const unsigned* xsc = smem + buf * 512;
    const unsigned* wlc = smem + 1024 + buf * 4096;
    Frag a, f0, f1, f2, f3;
    a.q[0]  = *(const uint4*)(xsc + (rowTile * 16 + m) * 16 + hh * 4);
    a.q[1]  = *(const uint4*)(xsc + (rowTile * 16 + m) * 16 + hh * 4 + 8);
    int nb = (cg * 64 + m) * 16 + hh * 4;
    f0.q[0] = *(const uint4*)(wlc + nb);
    f0.q[1] = *(const uint4*)(wlc + nb + 8);
    f1.q[0] = *(const uint4*)(wlc + nb + 256);
    f1.q[1] = *(const uint4*)(wlc + nb + 256 + 8);
    f2.q[0] = *(const uint4*)(wlc + nb + 512);
    f2.q[1] = *(const uint4*)(wlc + nb + 512 + 8);
    f3.q[0] = *(const uint4*)(wlc + nb + 768);
    f3.q[1] = *(const uint4*)(wlc + nb + 768 + 8);
    acc0 = __builtin_amdgcn_wmma_f32_16x16x32_bf16(false, a.v, false, f0.v, (short)0, acc0, false, false);
    acc1 = __builtin_amdgcn_wmma_f32_16x16x32_bf16(false, a.v, false, f1.v, (short)0, acc1, false, false);
    acc2 = __builtin_amdgcn_wmma_f32_16x16x32_bf16(false, a.v, false, f2.v, (short)0, acc2, false, false);
    acc3 = __builtin_amdgcn_wmma_f32_16x16x32_bf16(false, a.v, false, f3.v, (short)0, acc3, false, false);
  }
  __syncthreads();   // retire GEMM-phase LDS before aliasing

  float*    hs  = (float*)smem;           // 32 x 256
  unsigned* hnp = smem + 8192;            // 32 x 128 (bf16 pairs)
  float*    lgp = (float*)(smem + 12288); // 32 x 64

#pragma unroll
  for (int j = 0; j < 4; ++j) {
    v8f ar = (j == 0 ? acc0 : j == 1 ? acc1 : j == 2 ? acc2 : acc3);
#pragma unroll
    for (int v = 0; v < 8; ++v)
      hs[(rowTile * 16 + v + hh * 8) * 256 + cg * 64 + j * 16 + m] = ar[v];
  }
  __syncthreads();

  // LayerNorm(256) + L2 norm; 8 lanes per row
  {
    int r = tid >> 3, g = tid & 7;
    float vbuf[32]; float s = 0.f, q = 0.f;
#pragma unroll
    for (int i = 0; i < 32; ++i) {
      int c = g * 32 + i;
      float v = hs[r * 256 + c] + b1[c];
      vbuf[i] = v; s += v; q += v * v;
    }
#pragma unroll
    for (int mm = 1; mm < 8; mm <<= 1) { s += __shfl_xor(s, mm, 32); q += __shfl_xor(q, mm, 32); }
    float mu = s * (1.f / 256.f);
    float rstd = rsqrtf(q * (1.f / 256.f) - mu * mu + 1e-5f);
    float ss = 0.f;
#pragma unroll
    for (int i = 0; i < 32; ++i) {
      int c = g * 32 + i;
      float hv = (vbuf[i] - mu) * rstd * g1[c] + be1[c];
      vbuf[i] = hv; ss += hv * hv;
    }
#pragma unroll
    for (int mm = 1; mm < 8; mm <<= 1) ss += __shfl_xor(ss, mm, 32);
    float inv = 1.f / fmaxf(sqrtf(ss), 1e-12f);
#pragma unroll
    for (int i = 0; i < 16; ++i)
      hnp[r * 128 + g * 16 + i] = pack2(vbuf[2 * i] * inv, vbuf[2 * i + 1] * inv);
  }
  __syncthreads();

  // logits = h_norm @ cb_norm^T : each wave one 16x16 tile, K=256
  {
    v8f al = zero8();
    const unsigned* cb = cbn32 + (size_t)bIdx * 8192;   // b * K2*BDIM/2
    int n = (cg * 16 + m) * 16 + hh * 4;
#pragma unroll
    for (int step = 0; step < 8; ++step) {
      Frag a, bf;
      a.q[0]  = *(const uint4*)(hnp + (rowTile * 16 + m) * 128 + step * 16 + hh * 4);
      a.q[1]  = *(const uint4*)(hnp + (rowTile * 16 + m) * 128 + step * 16 + hh * 4 + 8);
      bf.q[0] = *(const uint4*)(cb + step * 1024 + n);
      bf.q[1] = *(const uint4*)(cb + step * 1024 + n + 8);
      al = __builtin_amdgcn_wmma_f32_16x16x32_bf16(false, a.v, false, bf.v, (short)0, al, false, false);
    }
#pragma unroll
    for (int v = 0; v < 8; ++v)
      lgp[(rowTile * 16 + v + hh * 8) * 64 + cg * 16 + m] = al[v];
  }
  __syncthreads();

  // softmax + first-match argmax + emb_low gather (8 lanes per row)
  {
    int r = tid >> 3, g = tid & 7;
    size_t rowg = row0 + r;
    float lv[8]; float mx = -3.4e38f; int mi = 0;
#pragma unroll
    for (int i = 0; i < 8; ++i) {
      int k = g * 8 + i;
      float v = lgp[r * 64 + k];
      lv[i] = v;
      if (v > mx) { mx = v; mi = k; }
    }
#pragma unroll
    for (int mm = 1; mm < 8; mm <<= 1) {
      float om = __shfl_xor(mx, mm, 32);
      int   oi = __shfl_xor(mi, mm, 32);
      if (om > mx || (om == mx && oi < mi)) { mx = om; mi = oi; }
    }
    float se = 0.f;
#pragma unroll
    for (int i = 0; i < 8; ++i) se += __expf(lv[i] - mx);
#pragma unroll
    for (int mm = 1; mm < 8; mm <<= 1) se += __shfl_xor(se, mm, 32);
    float rse = 1.f / se;
#pragma unroll
    for (int i = 0; i < 8; ++i) out_soft[rowg * KK2 + g * 8 + i] = __expf(lv[i] - mx) * rse;
    if (g == 0) out_hard[rowg] = (float)mi;
    const float* crow = cbf + ((size_t)bIdx * KK2 + mi) * BDIM;
#pragma unroll
    for (int i = 0; i < 32; ++i) out_low[rowg * BDIM + g * 32 + i] = crow[g * 32 + i];
  }
}

// ---- GEMM2 + LayerNorm(1024) fully in registers ----
__global__ __launch_bounds__(256) void project(
    const float* __restrict__ emb_low, const unsigned* __restrict__ w2t32,
    const float* __restrict__ b2, const float* __restrict__ g2, const float* __restrict__ be2,
    float* __restrict__ out_emb) {
  __shared__ __align__(16) unsigned as[32 * 128];   // A tile bf16 pairs (32 x 256)
  __shared__ float ssum[32 * 4], ssq[32 * 4], smu[32], srs[32];

  int tid = threadIdx.x, lane = tid & 31, wave = tid >> 5;
  int rowTile = wave >> 2, cg = wave & 3;
  int m = lane & 15, hh = lane >> 4;
  size_t row0 = (size_t)blockIdx.x * 32;

  {
    int r = tid >> 3, g = tid & 7;
    const float4* src = (const float4*)(emb_low + (row0 + r) * BDIM + g * 32);
#pragma unroll
    for (int i = 0; i < 8; ++i) {
      float4 f = src[i];
      as[r * 128 + g * 16 + i * 2]     = pack2(f.x, f.y);
      as[r * 128 + g * 16 + i * 2 + 1] = pack2(f.z, f.w);
    }
  }
  __syncthreads();

  v8f acc[16];
#pragma unroll
  for (int j = 0; j < 16; ++j) acc[j] = zero8();

  for (int step = 0; step < 8; ++step) {
    Frag a;
    a.q[0] = *(const uint4*)(as + (rowTile * 16 + m) * 128 + step * 16 + hh * 4);
    a.q[1] = *(const uint4*)(as + (rowTile * 16 + m) * 128 + step * 16 + hh * 4 + 8);
    const unsigned* wp = w2t32 + (size_t)step * 16384 + (size_t)(cg * 256 + m) * 16 + hh * 4;
    Frag bc, bn;   // even/odd 2-deep pipeline, named regs only
    bc.q[0] = *(const uint4*)(wp);
    bc.q[1] = *(const uint4*)(wp + 8);
#pragma unroll
    for (int j = 0; j < 16; j += 2) {
      if (j + 1 < 16) {
        const unsigned* p = wp + (j + 1) * 256;
        bn.q[0] = *(const uint4*)p;
        bn.q[1] = *(const uint4*)(p + 8);
      }
      acc[j] = __builtin_amdgcn_wmma_f32_16x16x32_bf16(false, a.v, false, bc.v, (short)0, acc[j], false, false);
      if (j + 2 < 16) {
        const unsigned* p = wp + (j + 2) * 256;
        bc.q[0] = *(const uint4*)p;
        bc.q[1] = *(const uint4*)(p + 8);
      }
      acc[j + 1] = __builtin_amdgcn_wmma_f32_16x16x32_bf16(false, a.v, false, bn.v, (short)0, acc[j + 1], false, false);
    }
  }
#pragma unroll
  for (int j = 0; j < 16; ++j) {
    float bias = b2[cg * 256 + j * 16 + m];
#pragma unroll
    for (int v = 0; v < 8; ++v) acc[j][v] += bias;
  }
  // per-row partial stats across 16 lanes of each half and this wave's 256 cols
#pragma unroll
  for (int v = 0; v < 8; ++v) {
    float s = 0.f, q = 0.f;
#pragma unroll
    for (int j = 0; j < 16; ++j) { float val = acc[j][v]; s += val; q += val * val; }
#pragma unroll
    for (int mm = 1; mm < 16; mm <<= 1) { s += __shfl_xor(s, mm, 32); q += __shfl_xor(q, mm, 32); }
    if (m == 0) {
      int r = rowTile * 16 + v + hh * 8;
      ssum[r * 4 + cg] = s; ssq[r * 4 + cg] = q;
    }
  }
  __syncthreads();
  if (tid < 32) {
    float s = ssum[tid * 4] + ssum[tid * 4 + 1] + ssum[tid * 4 + 2] + ssum[tid * 4 + 3];
    float q = ssq[tid * 4] + ssq[tid * 4 + 1] + ssq[tid * 4 + 2] + ssq[tid * 4 + 3];
    float mu = s * (1.f / 1024.f);
    smu[tid] = mu;
    srs[tid] = rsqrtf(q * (1.f / 1024.f) - mu * mu + 1e-5f);
  }
  __syncthreads();
#pragma unroll
  for (int j = 0; j < 16; ++j) {
    int c = cg * 256 + j * 16 + m;
    float gv = g2[c], bv = be2[c];
#pragma unroll
    for (int v = 0; v < 8; ++v) {
      int r = rowTile * 16 + v + hh * 8;
      out_emb[(row0 + r) * DD + c] = (acc[j][v] - smu[r]) * srs[r] * gv + bv;
    }
  }
}

extern "C" void kernel_launch(void* const* d_in, const int* in_sizes, int n_in,
                              void* d_out, int out_size, void* d_ws, size_t ws_size,
                              hipStream_t stream) {
  (void)in_sizes; (void)n_in; (void)out_size; (void)ws_size;
  const float*     x    = (const float*)d_in[0];
  const long long* l1   = (const long long*)d_in[1];
  const float*     W1   = (const float*)d_in[2];
  const float*     b1   = (const float*)d_in[3];
  const float*     g1   = (const float*)d_in[4];
  const float*     be1  = (const float*)d_in[5];
  const float*     cbks = (const float*)d_in[6];
  const float*     W2   = (const float*)d_in[7];
  const float*     b2   = (const float*)d_in[8];
  const float*     g2   = (const float*)d_in[9];
  const float*     be2  = (const float*)d_in[10];

  char* ws = (char*)d_ws;
  float*          cbf = (float*)ws;                                          // 2 MB
  unsigned short* cbn = (unsigned short*)(ws + (size_t)BB * KK2 * BDIM * 4); // 1 MB
  unsigned short* w1t = (unsigned short*)(ws + (size_t)BB * KK2 * BDIM * 6); // 512 KB
  unsigned short* w2t = (unsigned short*)((char*)w1t + (size_t)DD * BDIM * 2);

  float* out      = (float*)d_out;
  float* out_hard = out;
  float* out_soft = out_hard + (size_t)BB * TT;
  float* out_emb  = out_soft + (size_t)BB * TT * KK2;
  float* out_low  = out_emb + (size_t)BB * TT * DD;

  convert_w1<<<(DD * BDIM) / 256, 256, 0, stream>>>(W1, w1t);
  convert_w2<<<(DD * BDIM) / 256, 256, 0, stream>>>(W2, w2t);
  prep_cb<<<BB, 256, 0, stream>>>(cbks, l1, cbf, cbn);
  bottleneck<<<(BB * TT) / 32, 256, 0, stream>>>(x, (const unsigned*)w1t, b1, g1, be1,
                                                 cbf, (const unsigned*)cbn,
                                                 out_hard, out_soft, out_low);
  project<<<(BB * TT) / 32, 256, 0, stream>>>(out_low, (const unsigned*)w2t, b2, g2, be2, out_emb);
}
